// SSIMLoss_37615323578619
// MI455X (gfx1250) — compile-verified, matrix operand order checked by reading, not run on hardware
//
#include <hip/hip_runtime.h>

// SSIM loss on [8,4,512,512] f32 via bf16 WMMA separable Gaussian conv.
// Memory floor: 64 MiB / 23.3 TB/s ~= 2.9 us. Conv MACs routed through
// v_wmma_f32_16x16x32_bf16 (15 WMMAs per 16x16 tile, 32768 tiles).

typedef __attribute__((ext_vector_type(16))) __bf16 v16bf;
typedef __attribute__((ext_vector_type(8)))  float  v8f;

#define IMG_H   512
#define IMG_W   512
#define NIMG    32          // 8*4
#define LDS_PITCH 144       // 128 block cols + 2*5 halo, padded to mult of 16B
#define NPIX    8388608.0f  // 8*4*512*512

__device__ inline v8f wmma_bf16(v16bf a, v16bf b, v8f c) {
  // 8 args: (neg_a, A, neg_b, B, c_mod, C, reuse_a, reuse_b)
  return __builtin_amdgcn_wmma_f32_16x16x32_bf16(false, a, false, b,
                                                 (short)0, c, false, false);
}

// Full separable 11-tap conv for one 16x16 tile:
//   h-pass: D1 = A1 * W (rows 0..15 of halo), D2 = A2 * W (rows 16..31)
//   v-pass: S  = W * pack(D1, D2)
// W's band zeros kill the K=26..31 padding in both passes.
// f32 C/D layout feeds the bf16 B operand lane-for-lane: bv[e]=D1[e], bv[8+e]=D2[e].
__device__ inline v8f conv_tile(v16bf a1, v16bf a2, v16bf w) {
  v8f z = {0.f, 0.f, 0.f, 0.f, 0.f, 0.f, 0.f, 0.f};
  v8f d1 = wmma_bf16(a1, w, z);
  v8f d2 = wmma_bf16(a2, w, z);
  v16bf bv;
#pragma unroll
  for (int v = 0; v < 8; ++v) {
    bv[v]     = (__bf16)d1[v];
    bv[8 + v] = (__bf16)d2[v];
  }
  return wmma_bf16(w, bv, z);
}

__global__ __launch_bounds__(256)
void ssim_map_kernel(const float* __restrict__ pred,
                     const float* __restrict__ targ,
                     float* __restrict__ partials) {
  __shared__ float lp[32 * LDS_PITCH];
  __shared__ float lt[32 * LDS_PITCH];
  __shared__ float red[256];

  const int tid  = threadIdx.x;
  const int img  = blockIdx.z;                 // 0..31 (b*4+c linear)
  const int r0   = blockIdx.y * 16;            // tile row strip
  const int cblk = blockIdx.x * 128;           // 8 tiles of 16 cols per block
  const size_t base = (size_t)img * IMG_H * IMG_W;

  // --- cooperative stage of 32x144 halo region (zero-padded SAME) ---
  for (int i = tid; i < 32 * LDS_PITCH; i += 256) {
    int row = i / LDS_PITCH;
    int col = i - row * LDS_PITCH;
    int gr = r0 - 5 + row;
    int gc = cblk - 5 + col;
    float pv = 0.f, tv = 0.f;
    if ((unsigned)gr < (unsigned)IMG_H && (unsigned)gc < (unsigned)IMG_W) {
      size_t idx = base + (size_t)gr * IMG_W + (size_t)gc;
      pv = pred[idx];
      tv = targ[idx];
    }
    lp[i] = pv;
    lt[i] = tv;
  }
  __syncthreads();

  const int wave    = tid >> 5;        // 8 waves -> 8 tiles of 16 cols
  const int lane    = tid & 31;
  const int x       = lane & 15;       // M index (A) / N index (B)
  const int hib     = (lane >> 4) << 3; // lanes>=16 hold K+8
  const int colbase = wave * 16;

  // Gaussian normalization (sigma=1.5, 11 taps)
  float ssum = 0.f;
#pragma unroll
  for (int i = 0; i < 11; ++i) {
    float d = (float)(i - 5);
    ssum += __expf(d * d * (-1.f / 4.5f));
  }
  const float ginv = 1.f / ssum;

  // Banded Gaussian matrix: identical per-lane contents as A (v-pass)
  // and as B (h-pass):  W[K, x] = g[K - x] for K-x in [0,10], else 0.
  v16bf wm;
#pragma unroll
  for (int e = 0; e < 16; ++e) {
    int K = (e < 8) ? (e + hib) : (e + 8 + hib);
    int k = K - x - 5;                 // window offset in [-5,5]
    float wv = (k >= -5 && k <= 5) ? __expf((float)(k * k) * (-1.f / 4.5f)) * ginv
                                   : 0.f;
    wm[e] = (__bf16)wv;
  }

  // --- gather raw p,t for the two 16-row A blocks (aligned 8-f32 runs) ---
  float p1[16], t1[16], p2[16], t2[16];
  {
    const float* sp1 = &lp[(x)      * LDS_PITCH + colbase];
    const float* st1 = &lt[(x)      * LDS_PITCH + colbase];
    const float* sp2 = &lp[(x + 16) * LDS_PITCH + colbase];
    const float* st2 = &lt[(x + 16) * LDS_PITCH + colbase];
#pragma unroll
    for (int e = 0; e < 16; ++e) {
      int K = (e < 8) ? (e + hib) : (e + 8 + hib);
      p1[e] = sp1[K]; t1[e] = st1[K];
      p2[e] = sp2[K]; t2[e] = st2[K];
    }
  }

  // Five conv fields; products formed in f32 before bf16 rounding.
  v16bf aP1, aP2, aT1, aT2, aPP1, aPP2, aTT1, aTT2, aPT1, aPT2;
#pragma unroll
  for (int e = 0; e < 16; ++e) {
    aP1[e]  = (__bf16)p1[e];            aP2[e]  = (__bf16)p2[e];
    aT1[e]  = (__bf16)t1[e];            aT2[e]  = (__bf16)t2[e];
    aPP1[e] = (__bf16)(p1[e] * p1[e]);  aPP2[e] = (__bf16)(p2[e] * p2[e]);
    aTT1[e] = (__bf16)(t1[e] * t1[e]);  aTT2[e] = (__bf16)(t2[e] * t2[e]);
    aPT1[e] = (__bf16)(p1[e] * t1[e]);  aPT2[e] = (__bf16)(p2[e] * t2[e]);
  }

  v8f Smu1 = conv_tile(aP1,  aP2,  wm);
  v8f Smu2 = conv_tile(aT1,  aT2,  wm);
  v8f Spp  = conv_tile(aPP1, aPP2, wm);
  v8f Stt  = conv_tile(aTT1, aTT2, wm);
  v8f Spt  = conv_tile(aPT1, aPT2, wm);

  // --- elementwise SSIM map + per-lane accumulation (8 pixels/lane) ---
  const float C1 = 1e-4f;   // 0.01^2
  const float C2 = 9e-4f;   // 0.03^2
  float acc = 0.f;
#pragma unroll
  for (int v = 0; v < 8; ++v) {
    float mu1  = Smu1[v], mu2 = Smu2[v];
    float mu1s = mu1 * mu1, mu2s = mu2 * mu2, m12 = mu1 * mu2;
    float s1   = Spp[v] - mu1s;
    float s2   = Stt[v] - mu2s;
    float s12  = Spt[v] - m12;
    float num  = (2.f * m12 + C1) * (2.f * s12 + C2);
    float den  = (mu1s + mu2s + C1) * (s1 + s2 + C2);
    acc += num / den;
  }

  // --- deterministic block reduction -> one partial per block ---
  red[tid] = acc;
  __syncthreads();
#pragma unroll
  for (int off = 128; off > 0; off >>= 1) {
    if (tid < off) red[tid] += red[tid + off];
    __syncthreads();
  }
  if (tid == 0) {
    partials[(blockIdx.z * gridDim.y + blockIdx.y) * gridDim.x + blockIdx.x] = red[0];
  }
}

__global__ __launch_bounds__(256)
void ssim_finalize_kernel(const float* __restrict__ partials, int n,
                          float* __restrict__ out) {
  __shared__ float red[256];
  float s = 0.f;
  for (int i = threadIdx.x; i < n; i += 256) s += partials[i];
  red[threadIdx.x] = s;
  __syncthreads();
#pragma unroll
  for (int off = 128; off > 0; off >>= 1) {
    if (threadIdx.x < off) red[threadIdx.x] += red[threadIdx.x + off];
    __syncthreads();
  }
  if (threadIdx.x == 0) out[0] = 1.f - red[0] * (1.f / NPIX);
}

extern "C" void kernel_launch(void* const* d_in, const int* in_sizes, int n_in,
                              void* d_out, int out_size, void* d_ws, size_t ws_size,
                              hipStream_t stream) {
  const float* pred = (const float*)d_in[0];
  const float* targ = (const float*)d_in[1];
  float* out      = (float*)d_out;
  float* partials = (float*)d_ws;     // 4096 f32 partials = 16 KB scratch

  dim3 grid(IMG_W / 128, IMG_H / 16, NIMG);   // (4, 32, 32) = 4096 blocks
  ssim_map_kernel<<<grid, 256, 0, stream>>>(pred, targ, partials);

  const int nblocks = (IMG_W / 128) * (IMG_H / 16) * NIMG;
  ssim_finalize_kernel<<<1, 256, 0, stream>>>(partials, nblocks, out);
}
